// BCM_Linear_75754633167435
// MI455X (gfx1250) — matrix operand order repaired and checked by reading, not compiled, not run on hardware
//
#include <hip/hip_runtime.h>
#include <hip/hip_bf16.h>
#include <stdint.h>

typedef __bf16 v16bf __attribute__((ext_vector_type(16)));
typedef __bf16 v8bf  __attribute__((ext_vector_type(8)));
typedef float  v8f   __attribute__((ext_vector_type(8)));
typedef float  v4f   __attribute__((ext_vector_type(4)));

#define N_TOK   32768
#define CH      1024      // IN_CHANNELS == OUT_CHANNELS
#define BM      128
#define BN      128
#define BK      32
#define LDSROW  40        // padded LDS row stride in halves (80B = 20 dwords, conflict-free)
#define KSTEPS  (CH / BK) // 32

// ---------------------------------------------------------------------------
// CDNA5 async global->LDS copy (per-lane 16B), tracked by ASYNCcnt.
// LDS destination operand = wave-relative LDS byte offset (low 32 bits of the
// generic pointer to a __shared__ object, per ISA 10.2 aperture truncation).
// ---------------------------------------------------------------------------
__device__ __forceinline__ void async_b128(uint32_t lds_off, const void* g) {
  asm volatile("global_load_async_to_lds_b128 %0, %1, off"
               :: "v"(lds_off), "v"((unsigned long long)(uintptr_t)g)
               : "memory");
}
__device__ __forceinline__ void wait_async0() {
  asm volatile("s_wait_asynccnt 0x0" ::: "memory");
}

// ---------------------------------------------------------------------------
// Kernel 1: materialize the block-circulant dense weight as bf16 hi/lo pair.
// W[r*64+i, q*64+j] = w[r, q, (i-j) & 63];  Wh = bf16(W), Wl = bf16(W - Wh).
// ---------------------------------------------------------------------------
__global__ __launch_bounds__(256) void bcm_build_w(
    const float* __restrict__ w, __bf16* __restrict__ wh, __bf16* __restrict__ wl) {
  int idx = blockIdx.x * 256 + threadIdx.x;   // 0 .. CH*CH-1
  int o = idx >> 10, k = idx & (CH - 1);
  int r = o >> 6,  i = o & 63;
  int q = k >> 6,  j = k & 63;
  float v = w[(r * 16 + q) * 64 + ((i - j) & 63)];
  __bf16 h = (__bf16)v;
  wh[idx] = h;
  wl[idx] = (__bf16)(v - (float)h);
}

__device__ inline v16bf cat8(v8bf a, v8bf b) {
  v16bf r;
#pragma unroll
  for (int i = 0; i < 8; ++i) { r[i] = a[i]; r[i + 8] = b[i]; }
  return r;
}

// ---------------------------------------------------------------------------
// Kernel 2: out = x @ W.T via bf16x3-split WMMA (fp32 accumulate).
// Block tile: 128(M) x 128(N), K-step 32. 8 waves in 4(M) x 2(N) grid,
// each wave computes a 32x32 C region (2x2 WMMA tiles, 3 WMMAs per tile-K).
// A (x, fp32) goes global->regs->split->LDS; B (Wh/Wl, bf16) goes
// global->LDS via GLOBAL_LOAD_ASYNC_TO_LDS_B128 (ASYNCcnt).
// ---------------------------------------------------------------------------
__global__ __launch_bounds__(256) void bcm_gemm(
    const float* __restrict__ x, const __bf16* __restrict__ wh,
    const __bf16* __restrict__ wl, float* __restrict__ out) {
  __shared__ __bf16 sAh[2][BM * LDSROW];
  __shared__ __bf16 sAl[2][BM * LDSROW];
  __shared__ __bf16 sBh[2][BN * LDSROW];
  __shared__ __bf16 sBl[2][BN * LDSROW];

  const int t    = threadIdx.x;
  const int lane = t & 31;
  const int wv   = t >> 5;
  const int wm   = wv & 3;        // wave M index: 0..3  (32 rows each)
  const int wn   = wv >> 2;       // wave N index: 0..1  (64 cols each)
  const int lh   = lane >> 4;     // lane half-group (0/1)
  const int ln   = lane & 15;
  const int n0   = blockIdx.x * BN;   // N fastest-varying: 8 blocks share one
  const int m0   = blockIdx.y * BM;   // x panel -> x stays L2-resident

  // Cooperative loader: thread t handles 16 contiguous K elements of one row.
  const int lrow = t >> 1;         // 0..127 (M row for A, N row for B)
  const int lk   = (t & 1) * 16;   // 0 or 16

  v8f acc[2][2];
#pragma unroll
  for (int mi = 0; mi < 2; ++mi)
#pragma unroll
    for (int ni = 0; ni < 2; ++ni)
#pragma unroll
      for (int e = 0; e < 8; ++e) acc[mi][ni][e] = 0.0f;

  // A prefetch registers (global -> regs -> split -> LDS)
  v4f a_f[4];

  auto async_B = [&](int ks, int buf) {
    const __bf16* bhp = wh + (size_t)(n0 + lrow) * CH + ks * BK + lk;
    const __bf16* blp = wl + (size_t)(n0 + lrow) * CH + ks * BK + lk;
    const int ro = lrow * LDSROW + lk;
    async_b128((uint32_t)(uintptr_t)&sBh[buf][ro],     bhp);
    async_b128((uint32_t)(uintptr_t)&sBh[buf][ro + 8], bhp + 8);
    async_b128((uint32_t)(uintptr_t)&sBl[buf][ro],     blp);
    async_b128((uint32_t)(uintptr_t)&sBl[buf][ro + 8], blp + 8);
  };

  auto g_loadA = [&](int ks) {
    const float* ap = x + (size_t)(m0 + lrow) * CH + ks * BK + lk;
#pragma unroll
    for (int u = 0; u < 4; ++u) a_f[u] = ((const v4f*)ap)[u];
  };

  auto lds_storeA = [&](int buf) {
    v8bf ah[2], al[2];
#pragma unroll
    for (int u = 0; u < 4; ++u)
#pragma unroll
      for (int e = 0; e < 4; ++e) {
        float f  = a_f[u][e];
        __bf16 h = (__bf16)f;
        ah[u >> 1][(u & 1) * 4 + e] = h;
        al[u >> 1][(u & 1) * 4 + e] = (__bf16)(f - (float)h);
      }
    const int ro = lrow * LDSROW + lk;
    *(v8bf*)&sAh[buf][ro]     = ah[0];
    *(v8bf*)&sAh[buf][ro + 8] = ah[1];
    *(v8bf*)&sAl[buf][ro]     = al[0];
    *(v8bf*)&sAl[buf][ro + 8] = al[1];
  };

  // Prologue: fill buffer 0.
  async_B(0, 0);
  g_loadA(0);
  lds_storeA(0);
  wait_async0();
  __syncthreads();

  for (int ks = 0; ks < KSTEPS; ++ks) {
    const int cur = ks & 1;
    if (ks + 1 < KSTEPS) {
      async_B(ks + 1, cur ^ 1);  // DMA B panel for next step
      g_loadA(ks + 1);           // overlap A global latency with WMMA
    }

#pragma unroll
    for (int mi = 0; mi < 2; ++mi) {
      // A fragment: row M = ln, 16 contiguous K at 16*lh
      const int ar = (wm * 32 + mi * 16 + ln) * LDSROW;
      v16bf Ah = cat8(*(const v8bf*)&sAh[cur][ar + lh * 16],
                      *(const v8bf*)&sAh[cur][ar + lh * 16 + 8]);
      v16bf Al = cat8(*(const v8bf*)&sAl[cur][ar + lh * 16],
                      *(const v8bf*)&sAl[cur][ar + lh * 16 + 8]);
#pragma unroll
      for (int ni = 0; ni < 2; ++ni) {
        // B fragment: col N = ln, 16 contiguous K at 16*lh  (same k order as A)
        const int br = (wn * 64 + ni * 16 + ln) * LDSROW + lh * 16;
        v16bf Bh = cat8(*(const v8bf*)&sBh[cur][br],
                        *(const v8bf*)&sBh[cur][br + 8]);
        v16bf Bl = cat8(*(const v8bf*)&sBl[cur][br],
                        *(const v8bf*)&sBl[cur][br + 8]);
        // bf16x3: Ah*Bh + Ah*Bl + Al*Bh, fp32 accumulate
        acc[mi][ni] = __builtin_amdgcn_wmma_f32_16x16x32_bf16(
            false, Ah, false, Bh, (short)0, acc[mi][ni], false, false);
        acc[mi][ni] = __builtin_amdgcn_wmma_f32_16x16x32_bf16(
            false, Ah, false, Bl, (short)0, acc[mi][ni], false, false);
        acc[mi][ni] = __builtin_amdgcn_wmma_f32_16x16x32_bf16(
            false, Al, false, Bh, (short)0, acc[mi][ni], false, false);
      }
    }

    if (ks + 1 < KSTEPS) lds_storeA(cur ^ 1);
    wait_async0();     // B DMA for next buffer complete before barrier
    __syncthreads();
  }

  // C/D layout: VGPR v, lane L -> M = v + 8*(L/16), N = L%16
#pragma unroll
  for (int mi = 0; mi < 2; ++mi)
#pragma unroll
    for (int ni = 0; ni < 2; ++ni) {
      const int col = n0 + wn * 64 + ni * 16 + ln;
#pragma unroll
      for (int vv = 0; vv < 8; ++vv) {
        const int row = m0 + wm * 32 + mi * 16 + vv + lh * 8;
        out[(size_t)row * CH + col] = acc[mi][ni][vv];
      }
    }
}

// ---------------------------------------------------------------------------
extern "C" void kernel_launch(void* const* d_in, const int* in_sizes, int n_in,
                              void* d_out, int out_size, void* d_ws, size_t ws_size,
                              hipStream_t stream) {
  const float* x = (const float*)d_in[0];   // (32768, 1024) fp32
  const float* w = (const float*)d_in[1];   // (16, 16, 64) fp32
  float* out = (float*)d_out;               // (32768, 1024) fp32

  // Workspace: Wh (2 MB) + Wl (2 MB) bf16 dense weights.
  __bf16* whp = (__bf16*)d_ws;
  __bf16* wlp = whp + (size_t)CH * CH;

  bcm_build_w<<<(CH * CH) / 256, 256, 0, stream>>>(w, whp, wlp);
  bcm_gemm<<<dim3(CH / BN, N_TOK / BM), 256, 0, stream>>>(x, whp, wlp, out);
}
